// Decoder_2_44805098831970
// MI455X (gfx1250) — matrix operand order repaired and checked by reading, not compiled, not run on hardware
//
#include <hip/hip_runtime.h>

typedef __attribute__((ext_vector_type(16))) __bf16 bf16x16;
typedef __attribute__((ext_vector_type(8)))  float  f32x8;
typedef unsigned short u16;

#define ACT_NONE 0
#define ACT_RELU 1
#define ACT_GELU 2
#define ACT_SIG  3

__device__ __forceinline__ u16 f2bf(float f) {
    unsigned u = __float_as_uint(f);
    u += 0x7FFFu + ((u >> 16) & 1u);
    return (u16)(u >> 16);
}

// ---------------- layout conversion: fp32 NCHW -> bf16 NHWC (C padded to Cp) ----------
__global__ __launch_bounds__(256) void k_nhwc(const float* __restrict__ src,
                                              u16* __restrict__ dst,
                                              int C, int Cp, int H, int W, unsigned total) {
    unsigned idx = blockIdx.x * 256u + threadIdx.x;
    if (idx >= total) return;
    int c = idx % Cp; unsigned p = idx / Cp;
    int w = p % W; p /= W; int h = p % H; int b = p / H;
    float v = 0.f;
    if (c < C) v = src[(((size_t)b * C + c) * H + h) * W + w];
    dst[idx] = f2bf(v);
}

// ------------- weight repack: OIHW fp32 -> B-fragment order bf16 ----------------------
// tile (tap t, kblock kb, nblock nb): 32 lanes x 16 bf16 contiguous.
// lane L: column N = nb*16 + (L&15); K = kb*32 + (L>>4)*8 + (e&7) + (e>>3)*16, e=0..15.
__global__ __launch_bounds__(256) void k_packw(const float* __restrict__ w,
                                               u16* __restrict__ dst,
                                               int Cout, int Cin, int ks,
                                               int KB, int NBt, unsigned total) {
    unsigned idx = blockIdx.x * 256u + threadIdx.x;
    if (idx >= total) return;
    int e = idx & 15;
    int L = (idx >> 4) & 31;
    unsigned q = idx >> 9;
    int nb = q % NBt; q /= NBt;
    int kb = q % KB;
    int t  = q / KB;
    int N = nb * 16 + (L & 15);
    int K = kb * 32 + (L >> 4) * 8 + (e & 7) + (e >> 3) * 16;
    float v = 0.f;
    if (K < Cin) v = w[(((size_t)N * Cin + K) * ks + t / ks) * ks + (t % ks)];
    dst[idx] = f2bf(v);
}

// ------------- implicit-GEMM conv via v_wmma_f32_16x16x32_bf16 ------------------------
// One wave per 16-pixel row strip; wave holds ALL Cout/16 accumulator tiles and
// streams B fragments against a single A fragment per (tap, kblock).
// Grid: (W/16, H, batch), block = 32 threads (one wave32).
template <int H, int W, int Cp, int Cout, int KS, int DIL, int PAD>
__global__ __launch_bounds__(32) void k_conv(
    const u16* __restrict__ act, const u16* __restrict__ wt,
    const float* __restrict__ bias, const float* __restrict__ res,
    float* __restrict__ out, int actmode) {
    constexpr int KB  = Cp / 32;
    constexpr int NBt = Cout / 16;
    int lane = threadIdx.x & 31;
    int wsx = blockIdx.x;
    int h   = blockIdx.y;
    int b   = blockIdx.z;

    int M   = lane & 15;
    int hiK = lane >> 4;        // 0 or 1
    int ka0 = hiK * 8;          // A-fragment K chunk base (ISA 16-bit A layout)
    int wpix = (wsx << 4) + M;

    const f32x8 zero = {0.f, 0.f, 0.f, 0.f, 0.f, 0.f, 0.f, 0.f};
    f32x8 acc[NBt];
#pragma unroll
    for (int i = 0; i < NBt; ++i) acc[i] = zero;

#pragma unroll 1
    for (int kh = 0; kh < KS; ++kh) {
        int hh = h + kh * DIL - PAD;
        bool hok = (unsigned)hh < (unsigned)H;
#pragma unroll 1
        for (int kw = 0; kw < KS; ++kw) {
            int wwp = wpix + kw * DIL - PAD;
            bool ok = hok && ((unsigned)wwp < (unsigned)W);
            const u16* wtap = wt + (size_t)(kh * KS + kw) * KB * NBt * 512 + lane * 16;
#pragma unroll 1
            for (int kb = 0; kb < KB; ++kb) {
                uint4 a0 = make_uint4(0, 0, 0, 0), a1 = make_uint4(0, 0, 0, 0);
                if (ok) {
                    const u16* ab = act + (((size_t)b * H + hh) * W + wwp) * Cp + kb * 32 + ka0;
                    a0 = *(const uint4*)ab;          // K = ka0 .. ka0+7
                    a1 = *(const uint4*)(ab + 16);   // K = ka0+16 .. ka0+23
                }
                union { uint4 u[2]; bf16x16 v; } Af;
                Af.u[0] = a0; Af.u[1] = a1;
                const u16* wkb = wtap + (size_t)kb * NBt * 512;
                __builtin_prefetch(wkb + (size_t)NBt * 512, 0, 0);
#pragma unroll
                for (int nb = 0; nb < NBt; ++nb) {
                    union { uint4 u[2]; bf16x16 v; } Bf;
                    const uint4* wp = (const uint4*)(wkb + nb * 512);
                    Bf.u[0] = wp[0]; Bf.u[1] = wp[1];
                    acc[nb] = __builtin_amdgcn_wmma_f32_16x16x32_bf16(
                        false, Af.v, false, Bf.v, (short)0, acc[nb], false, false);
                }
            }
        }
    }

    // Epilogue: C/D layout -> lane%16 = column N, VGPR r -> row M = r + 8*(lane>=16)
#pragma unroll
    for (int nb = 0; nb < NBt; ++nb) {
        int n = (nb << 4) + M;
        float bv = bias ? bias[n] : 0.f;
#pragma unroll
        for (int ri = 0; ri < 8; ++ri) {
            int mr = ri + (hiK << 3);
            float v = acc[nb][ri] + bv;
            if (actmode == ACT_RELU)      v = fmaxf(v, 0.f);
            else if (actmode == ACT_GELU) v = 0.5f * v * (1.f + erff(v * 0.70710678118654752f));
            else if (actmode == ACT_SIG)  v = 1.f / (1.f + __expf(-v));
            size_t oi = (((size_t)b * Cout + n) * H + h) * W + ((wsx << 4) + mr);
            if (res) v += res[oi];
            out[oi] = v;
        }
    }
}

// ------------- mean over H*W per (b,c) ------------------------------------------------
__global__ __launch_bounds__(256) void k_mean(const float* __restrict__ x,
                                              float* __restrict__ pool, int HW) {
    int bc = blockIdx.x;
    const float* p = x + (size_t)bc * HW;
    float s = 0.f;
    for (int i = threadIdx.x; i < HW; i += 256) s += p[i];
    __shared__ float sm[256];
    sm[threadIdx.x] = s;
    __syncthreads();
    for (int o = 128; o > 0; o >>= 1) {
        if (threadIdx.x < o) sm[threadIdx.x] += sm[threadIdx.x + o];
        __syncthreads();
    }
    if (threadIdx.x == 0) pool[bc] = sm[0] / (float)HW;
}

// ------------- CA: relu(W1 @ pooled) -> sigmoid(W2 @ y1) per batch --------------------
__global__ __launch_bounds__(256) void k_ca(const float* __restrict__ pool,
                                            const float* __restrict__ w1,
                                            const float* __restrict__ w2,
                                            float* __restrict__ scl, int C) {
    int b = blockIdx.x, t = threadIdx.x, Cr = C / 4;
    __shared__ float pl[192];
    __shared__ float y1[48];
    if (t < C) pl[t] = pool[b * C + t];
    __syncthreads();
    if (t < Cr) {
        float s = 0.f;
        for (int c = 0; c < C; c++) s += w1[(size_t)t * C + c] * pl[c];
        y1[t] = fmaxf(s, 0.f);
    }
    __syncthreads();
    if (t < C) {
        float s = 0.f;
        for (int j = 0; j < Cr; j++) s += w2[(size_t)t * Cr + j] * y1[j];
        scl[b * C + t] = 1.f / (1.f + expf(-s));
    }
}

// ------------- out = t2 * scale[b,c] + x ----------------------------------------------
__global__ __launch_bounds__(256) void k_scale_res(const float* __restrict__ t2,
                                                   const float* __restrict__ scl,
                                                   const float* __restrict__ x,
                                                   float* __restrict__ out,
                                                   int HW, unsigned total) {
    unsigned idx = blockIdx.x * 256u + threadIdx.x;
    if (idx >= total) return;
    unsigned bc = idx / HW;
    out[idx] = t2[idx] * scl[bc] + x[idx];
}

__global__ __launch_bounds__(256) void k_mul(const float* __restrict__ a,
                                             const float* __restrict__ b,
                                             float* __restrict__ o, unsigned total) {
    unsigned idx = blockIdx.x * 256u + threadIdx.x;
    if (idx >= total) return;
    o[idx] = a[idx] * b[idx];
}

// ------------- depthwise 3x3 conv, stride 8, pad 2, dil 2 ----------------------------
__global__ __launch_bounds__(256) void k_dwconv(const float* __restrict__ in,
                                                const float* __restrict__ w,
                                                const float* __restrict__ bias,
                                                float* __restrict__ out,
                                                int C, int H, int W, int S, unsigned total) {
    unsigned idx = blockIdx.x * 256u + threadIdx.x;
    if (idx >= total) return;
    int wo = idx % S; unsigned p = idx / S;
    int ho = p % S; p /= S;
    int c = p % C; int b = p / C;
    float s = bias[c];
    const float* ip = in + (((size_t)b * C + c) * H) * W;
    for (int kh = 0; kh < 3; kh++) {
        int hi = ho * 8 - 2 + kh * 2;
        if (hi < 0 || hi >= H) continue;
        for (int kw = 0; kw < 3; kw++) {
            int wi = wo * 8 - 2 + kw * 2;
            if (wi < 0 || wi >= W) continue;
            s += ip[(size_t)hi * W + wi] * w[c * 9 + kh * 3 + kw];
        }
    }
    out[idx] = s;
}

// ------------- 8x8 block inverse-DCT: Re(ifft_ortho) row+col == cosine transform ------
__global__ __launch_bounds__(256) void k_iddct(const float* __restrict__ in,
                                               float* __restrict__ out,
                                               int S, unsigned total) {
    unsigned idx = blockIdx.x * 256u + threadIdx.x;
    if (idx >= total) return;
    int w = idx % S; unsigned p = idx / S;
    int h = p % S; int bc = p / S;
    int i = h & 7, j = w & 7, hb = h & ~7, wb = w & ~7;
    const float inv = 0.35355339059327373f;  // 1/sqrt(8)
    float ci[8], cj[8];
#pragma unroll
    for (int m = 0; m < 8; m++) {
        ci[m] = __cosf(0.78539816339744831f * (float)(i * m)) * inv;
        cj[m] = __cosf(0.78539816339744831f * (float)(j * m)) * inv;
    }
    const float* src = in + ((size_t)bc * S + hb) * S + wb;
    float s = 0.f;
#pragma unroll
    for (int m = 0; m < 8; m++) {
        float r = 0.f;
        const float* row = src + (size_t)m * S;
#pragma unroll
        for (int nn = 0; nn < 8; nn++) r += cj[nn] * row[nn];
        s += ci[m] * r;
    }
    out[idx] = s;
}

// ------------- bilinear (half-pixel, edge clamp) --------------------------------------
__device__ __forceinline__ void bilc(int o, float scale, int isz, int& i0, int& i1, float& f) {
    float x = ((float)o + 0.5f) * scale - 0.5f;
    x = fminf(fmaxf(x, 0.f), (float)(isz - 1));
    i0 = (int)x;
    i1 = (i0 + 1 < isz) ? i0 + 1 : isz - 1;
    f = x - (float)i0;
}

__global__ __launch_bounds__(256) void k_bil(const float* __restrict__ in,
                                             float* __restrict__ out,
                                             int Hi, int Wi, int Ho, int Wo, unsigned total) {
    unsigned idx = blockIdx.x * 256u + threadIdx.x;
    if (idx >= total) return;
    int w = idx % Wo; unsigned p = idx / Wo;
    int h = p % Ho; int bc = p / Ho;
    int y0, y1, x0, x1; float fy, fx;
    bilc(h, (float)Hi / Ho, Hi, y0, y1, fy);
    bilc(w, (float)Wi / Wo, Wi, x0, x1, fx);
    const float* s = in + (size_t)bc * Hi * Wi;
    float v00 = s[(size_t)y0 * Wi + x0], v01 = s[(size_t)y0 * Wi + x1];
    float v10 = s[(size_t)y1 * Wi + x0], v11 = s[(size_t)y1 * Wi + x1];
    out[idx] = (v00 * (1.f - fx) + v01 * fx) * (1.f - fy) + (v10 * (1.f - fx) + v11 * fx) * fy;
}

// fused: bilinear upsample (NCHW fp32) directly into NHWC bf16 staging
__global__ __launch_bounds__(256) void k_bil_nhwc(const float* __restrict__ in,
                                                  u16* __restrict__ out,
                                                  int C, int Cp, int Hi, int Wi,
                                                  int Ho, int Wo, unsigned total) {
    unsigned idx = blockIdx.x * 256u + threadIdx.x;
    if (idx >= total) return;
    int c = idx % Cp; unsigned p = idx / Cp;
    int w = p % Wo; p /= Wo;
    int h = p % Ho; int b = p / Ho;
    float v = 0.f;
    if (c < C) {
        int y0, y1, x0, x1; float fy, fx;
        bilc(h, (float)Hi / Ho, Hi, y0, y1, fy);
        bilc(w, (float)Wi / Wo, Wi, x0, x1, fx);
        const float* s = in + (((size_t)b * C + c) * Hi) * Wi;
        float v00 = s[(size_t)y0 * Wi + x0], v01 = s[(size_t)y0 * Wi + x1];
        float v10 = s[(size_t)y1 * Wi + x0], v11 = s[(size_t)y1 * Wi + x1];
        v = (v00 * (1.f - fx) + v01 * fx) * (1.f - fy) + (v10 * (1.f - fx) + v11 * fx) * fy;
    }
    out[idx] = f2bf(v);
}

// =============================== host-side helpers ====================================
static void nhwc_stage(hipStream_t s, const float* in, u16* ACT, int C, int Cp, int H, int W) {
    unsigned nA = (unsigned)((size_t)4 * H * W * Cp);
    k_nhwc<<<(nA + 255) / 256, 256, 0, s>>>(in, ACT, C, Cp, H, W, nA);
}

template <int H, int W, int Cp, int Cout, int KS, int DIL, int PAD>
static void conv_t(hipStream_t s, const u16* ACT, const float* w, u16* WT,
                   const float* bias, const float* res, float* out, int act, int Cin) {
    constexpr int KB = Cp / 32, NBt = Cout / 16;
    unsigned nW = (unsigned)(KS * KS * KB * NBt * 512);
    k_packw<<<(nW + 255) / 256, 256, 0, s>>>(w, WT, Cout, Cin, KS, KB, NBt, nW);
    dim3 g(W / 16, H, 4);
    k_conv<H, W, Cp, Cout, KS, DIL, PAD><<<g, 32, 0, s>>>(ACT, WT, bias, res, out, act);
}

template <int H, int W, int Cp, int C, int DIL>
static void dcab_t(hipStream_t s, const float* x, float* t1, float* t2, float* out,
                   const float* w1, const float* w2, const float* ca1, const float* ca2,
                   u16* ACT, u16* WT, float* POOL, float* SCL) {
    nhwc_stage(s, x, ACT, C, Cp, H, W);
    conv_t<H, W, Cp, C, 3, DIL, DIL>(s, ACT, w1, WT, nullptr, nullptr, t1, ACT_RELU, C);
    nhwc_stage(s, t1, ACT, C, Cp, H, W);
    conv_t<H, W, Cp, C, 3, DIL, DIL>(s, ACT, w2, WT, nullptr, nullptr, t2, ACT_NONE, C);
    k_mean<<<4 * C, 256, 0, s>>>(t2, POOL, H * W);
    k_ca<<<4, 256, 0, s>>>(POOL, ca1, ca2, SCL, C);
    unsigned n = (unsigned)((size_t)4 * C * H * W);
    k_scale_res<<<(n + 255) / 256, 256, 0, s>>>(t2, SCL, x, out, H * W, n);
}

template <int H, int W, int Cp, int C>
static void fsm_t(hipStream_t s, const float* x,
                  const float* conv_w, const float* conv_b,
                  const float* ddct_w, const float* ddct_b,
                  const float* dct_w, const float* dct_b,
                  const float* dw1_w, const float* dw1_b,
                  const float* dw3_w, const float* dw3_b,
                  const float* ar_w, const float* ar_b,
                  float* A, float* Bb, float* Cc, float* S0, float* S1,
                  u16* ACT, u16* WT) {
    nhwc_stage(s, x, ACT, C, Cp, H, W);
    conv_t<H, W, Cp, C, 3, 1, 1>(s, ACT, conv_w, WT, conv_b, nullptr, A, ACT_GELU, C);
    nhwc_stage(s, A, ACT, C, Cp, H, W);
    conv_t<H, W, Cp, C, 3, 1, 1>(s, ACT, ddct_w, WT, ddct_b, A, Bb, ACT_GELU, C);
    constexpr int S = H / 8;
    unsigned n0 = (unsigned)(4 * C * S * S);
    k_dwconv<<<(n0 + 255) / 256, 256, 0, s>>>(Bb, dct_w, dct_b, S0, C, H, W, S, n0);
    k_iddct<<<(n0 + 255) / 256, 256, 0, s>>>(S0, S1, S, n0);
    unsigned n1 = (unsigned)((size_t)4 * C * H * W);
    k_bil<<<(n1 + 255) / 256, 256, 0, s>>>(S1, A, S, S, H, W, n1);
    nhwc_stage(s, x, ACT, C, Cp, H, W);
    conv_t<H, W, Cp, C, 3, 1, 1>(s, ACT, dw1_w, WT, dw1_b, nullptr, Bb, ACT_RELU, C);
    nhwc_stage(s, Bb, ACT, C, Cp, H, W);
    conv_t<H, W, Cp, C, 1, 1, 0>(s, ACT, dw3_w, WT, dw3_b, nullptr, Cc, ACT_SIG, C);
    k_mul<<<(n1 + 255) / 256, 256, 0, s>>>(A, Cc, Bb, n1);
    nhwc_stage(s, Bb, ACT, C, Cp, H, W);
    conv_t<H, W, Cp, C, 3, 1, 1>(s, ACT, ar_w, WT, ar_b, nullptr, Cc, ACT_NONE, C);
}

template <int Hi, int Wi, int H, int W, int Cp, int Cin, int Cout>
static void skip_t(hipStream_t s, const float* inLow, const float* w,
                   const float* res, float* out, u16* ACT, u16* WT) {
    unsigned nA = (unsigned)((size_t)4 * H * W * Cp);
    k_bil_nhwc<<<(nA + 255) / 256, 256, 0, s>>>(inLow, ACT, Cin, Cp, Hi, Wi, H, W, nA);
    conv_t<H, W, Cp, Cout, 1, 1, 0>(s, ACT, w, WT, nullptr, res, out, ACT_NONE, Cin);
}

// ====================================== entry ========================================
extern "C" void kernel_launch(void* const* d_in, const int* in_sizes, int n_in,
                              void* d_out, int out_size, void* d_ws, size_t ws_size,
                              hipStream_t stream) {
    (void)in_sizes; (void)n_in; (void)out_size; (void)ws_size;
    const float* enc1 = (const float*)d_in[0];  // (4,80,256,256)
    const float* enc2 = (const float*)d_in[1];  // (4,128,128,128)
    const float* enc3 = (const float*)d_in[2];  // (4,176,64,64)
    auto P = [&](int i) { return (const float*)d_in[i]; };
    // params flatten order (sorted dict keys):
    // dec1[0]: 3 ca1, 4 ca2, 5 w1, 6 w2 ; dec1[1]: 7..10
    // dec2[0]: 11..14 ; dec2[1]: 15..18 ; dec3[0]: 19..22 ; dec3[1]: 23..26
    // fsm1: 27 ar_b,28 ar_w,29 conv_b,30 conv_w,31 dct_b,32 dct_w,33 ddct_b,34 ddct_w,
    //       35 dw1_b,36 dw1_w,37 dw3_b,38 dw3_w ; fsm2: 39..50 ; 51 up21_w ; 52 up32_w

    float* out1 = (float*)d_out;                          // dec1 (4,80,256,256)
    float* out2 = out1 + (size_t)4 * 80 * 256 * 256;      // dec2 (4,128,128,128)
    float* out3 = out2 + (size_t)4 * 128 * 128 * 128;     // dec3 (4,176,64,64)

    char* ws = (char*)d_ws;
    size_t off = 0;
    auto carve = [&](size_t bytes) -> char* {
        char* p = ws + off;
        off += (bytes + 255) & ~(size_t)255;
        return p;
    };
    const size_t BIG = (size_t)4 * 80 * 256 * 256 * sizeof(float);  // 83.9 MB
    float* A  = (float*)carve(BIG);
    float* Bb = (float*)carve(BIG);
    float* Cc = (float*)carve(BIG);
    u16* ACT  = (u16*)carve((size_t)4 * 256 * 256 * 128 * sizeof(u16));  // 67.1 MB
    u16* WT   = (u16*)carve((size_t)1 << 20);
    float* S0 = (float*)carve((size_t)4 * 80 * 32 * 32 * sizeof(float));
    float* S1 = (float*)carve((size_t)4 * 80 * 32 * 32 * sizeof(float));
    float* POOL = (float*)carve(4 * 192 * sizeof(float));
    float* SCL  = (float*)carve(4 * 192 * sizeof(float));

    // ---- dec3: 2x DCAB(dil=4) on enc3, C=176 (Cp=192) @64x64 ----
    dcab_t<64, 64, 192, 176, 4>(stream, enc3, Bb, Cc, A,
                                P(21), P(22), P(19), P(20), ACT, WT, POOL, SCL);
    dcab_t<64, 64, 192, 176, 4>(stream, A, Bb, Cc, out3,
                                P(25), P(26), P(23), P(24), ACT, WT, POOL, SCL);

    // ---- fsm2 on enc2 -> Cc, C=128 @128x128 ----
    fsm_t<128, 128, 128, 128>(stream, enc2,
                              P(42), P(41), P(46), P(45), P(44), P(43),
                              P(48), P(47), P(50), P(49), P(40), P(39),
                              A, Bb, Cc, S0, S1, ACT, WT);

    // ---- skip 3->2: bilinear(dec3) + 1x1 conv + fsm2 -> Bb ----
    skip_t<64, 64, 128, 128, 192, 176, 128>(stream, out3, P(52), Cc, Bb, ACT, WT);

    // ---- dec2: 2x DCAB(dil=2), C=128 @128x128 ----
    dcab_t<128, 128, 128, 128, 2>(stream, Bb, A, Cc, Cc,
                                  P(13), P(14), P(11), P(12), ACT, WT, POOL, SCL);
    dcab_t<128, 128, 128, 128, 2>(stream, Cc, A, Bb, out2,
                                  P(17), P(18), P(15), P(16), ACT, WT, POOL, SCL);

    // ---- fsm1 on enc1 -> Cc, C=80 (Cp=96) @256x256 ----
    fsm_t<256, 256, 96, 80>(stream, enc1,
                            P(30), P(29), P(34), P(33), P(32), P(31),
                            P(36), P(35), P(38), P(37), P(28), P(27),
                            A, Bb, Cc, S0, S1, ACT, WT);

    // ---- skip 2->1: bilinear(dec2) + 1x1 conv + fsm1 -> Bb ----
    skip_t<128, 128, 256, 256, 128, 128, 80>(stream, out2, P(51), Cc, Bb, ACT, WT);

    // ---- dec1: 2x DCAB(dil=1), C=80 @256x256 ----
    dcab_t<256, 256, 96, 80, 1>(stream, Bb, A, Cc, Cc,
                                P(5), P(6), P(3), P(4), ACT, WT, POOL, SCL);
    dcab_t<256, 256, 96, 80, 1>(stream, Cc, A, Bb, out1,
                                P(9), P(10), P(7), P(8), ACT, WT, POOL, SCL);
}
